// MoE_5231270166969
// MI455X (gfx1250) — matrix operand order repaired and checked by reading, not compiled, not run on hardware
//
#include <hip/hip_runtime.h>

#define T_DIM 2048
#define B_DIM 16
#define H_DIM 2048
#define E_DIM 8
#define R_DIM 8
#define TOPK  2
#define KR    16            // TOPK * R_DIM
#define XS_STRIDE 40        // padded LDS row stride (ushorts): 80B = 20 banks, conflict-free
#define TCHUNK 128          // pooling chunk along T
#define KSTEPS (H_DIM / 32) // 64 main K-iterations

typedef __attribute__((ext_vector_type(16))) __bf16 v16bf_t;
typedef __attribute__((ext_vector_type(8)))  float  v8f_t;

union V16BF { uint4 u[2]; v16bf_t v; };

#if defined(__has_builtin)
#if __has_builtin(__builtin_amdgcn_global_load_async_to_lds_b128) && \
    __has_builtin(__builtin_amdgcn_s_wait_asynccnt)
#define HAVE_ASYNC_LDS 1
#endif
#endif

#ifdef HAVE_ASYNC_LDS
typedef __attribute__((__vector_size__(4 * sizeof(int)))) int v4i_t;  // matches builtin param
#endif

__device__ __forceinline__ unsigned short f2bf(float f) {
  unsigned int u = __float_as_uint(f);
  u += 0x7FFFu + ((u >> 16) & 1u);      // round-to-nearest-even
  return (unsigned short)(u >> 16);
}

// Pack bf16(a) into low 16, bf16(b) into high 16.
// 2x v_add_nc_u32 (round-half-up) + 1x v_perm_b32.
__device__ __forceinline__ unsigned int pk_bf16(float a, float b) {
  unsigned int ua = __float_as_uint(a) + 0x8000u;
  unsigned int ub = __float_as_uint(b) + 0x8000u;
  return __builtin_amdgcn_perm(ub, ua, 0x07060302u);
}

// ---------------- pooling: two-stage deterministic mean over T ----------------
__global__ void __launch_bounds__(256)
pool_partial_kernel(const float* __restrict__ x, float* __restrict__ partial) {
  int col = blockIdx.x * 256 + threadIdx.x;       // 0 .. B*H-1
  int chunk = blockIdx.y;                         // 0 .. T/TCHUNK-1
  const float* p = x + (size_t)(chunk * TCHUNK) * (B_DIM * H_DIM) + col;
  float s = 0.f;
  for (int t = 0; t < TCHUNK; ++t) s += p[(size_t)t * (B_DIM * H_DIM)];
  partial[(size_t)chunk * (B_DIM * H_DIM) + col] = s;
}

__global__ void __launch_bounds__(256)
pool_reduce_kernel(const float* __restrict__ partial, float* __restrict__ pooled) {
  int col = blockIdx.x * 256 + threadIdx.x;
  float s = 0.f;
#pragma unroll
  for (int c = 0; c < T_DIM / TCHUNK; ++c)
    s += partial[(size_t)c * (B_DIM * H_DIM) + col];
  pooled[col] = s * (1.0f / T_DIM);
}

// ---------------- gating: logits, top-2, softmax ----------------
__global__ void gate_kernel(const float* __restrict__ pooled, const float* __restrict__ Wg,
                            int* __restrict__ sel, float* __restrict__ wgt) {
  __shared__ float logits[B_DIM][E_DIM];
  int lane = threadIdx.x & 31;
  int wave = threadIdx.x >> 5;                    // 0..7, each wave: 16 (b,e) pairs
  for (int p = wave * 16; p < wave * 16 + 16; ++p) {
    int b = p >> 3, e = p & 7;
    float s = 0.f;
    for (int i = lane; i < H_DIM; i += 32)
      s += pooled[b * H_DIM + i] * Wg[i * E_DIM + e];
    for (int off = 16; off; off >>= 1) s += __shfl_down(s, off, 32);
    if (lane == 0) logits[b][e] = s;
  }
  __syncthreads();
  if (threadIdx.x < B_DIM) {
    int b = threadIdx.x;
    float v[E_DIM];
    for (int e = 0; e < E_DIM; ++e) v[e] = logits[b][e];
    int e0 = 0;
    for (int e = 1; e < E_DIM; ++e) if (v[e] > v[e0]) e0 = e;   // ties -> lowest idx (jax)
    int e1 = -1;
    for (int e = 0; e < E_DIM; ++e) { if (e == e0) continue; if (e1 < 0 || v[e] > v[e1]) e1 = e; }
    float m  = fmaxf(v[e0], v[e1]);
    float z0 = __expf(v[e0] - m), z1 = __expf(v[e1] - m);
    float inv = 1.f / (z0 + z1);
    sel[b * TOPK + 0] = e0;  sel[b * TOPK + 1] = e1;
    wgt[b * TOPK + 0] = z0 * inv;  wgt[b * TOPK + 1] = z1 * inv;
  }
}

// ---------------- WbfT[n][k] = bf16(W[k][n]) (tiled transpose+convert) ----------------
__global__ void __launch_bounds__(256)
transposeW_kernel(const float* __restrict__ W, unsigned short* __restrict__ WbfT) {
  __shared__ unsigned short tile[32][33];
  int kt = blockIdx.x * 32;
  int nt = blockIdx.y * 32;
  int tn  = threadIdx.x & 31;
  int tk0 = threadIdx.x >> 5;
#pragma unroll
  for (int it = 0; it < 4; ++it) {
    int kk = tk0 + it * 8;
    tile[kk][tn] = f2bf(W[(size_t)(kt + kk) * H_DIM + nt + tn]);
  }
  __syncthreads();
  int tk  = threadIdx.x & 31;
  int tn0 = threadIdx.x >> 5;
#pragma unroll
  for (int it = 0; it < 4; ++it) {
    int nn = tn0 + it * 8;
    WbfT[(size_t)(nt + nn) * H_DIM + kt + tk] = tile[tk][nn];
  }
}

// ---------------- gather LoRA factors per batch ----------------
__global__ void __launch_bounds__(256)
factors_kernel(const float* __restrict__ A, const float* __restrict__ Bl,
               const int* __restrict__ sel, const float* __restrict__ wgt,
               unsigned short* __restrict__ AbfT, unsigned short* __restrict__ BwT) {
  int id = blockIdx.x * blockDim.x + threadIdx.x;   // 0 .. B*KR*H-1
  {
    int h  = id % H_DIM;
    int rp = (id / H_DIM) % KR;
    int b  = id / (H_DIM * KR);
    int kk = rp / R_DIM, r = rp % R_DIM;
    int e  = sel[b * TOPK + kk];
    AbfT[id] = f2bf(A[((size_t)e * H_DIM + h) * R_DIM + r]);
  }
  {
    int rp = id % KR;
    int n  = (id / KR) % H_DIM;
    int b  = id / (KR * H_DIM);
    int kk = rp / R_DIM, r = rp % R_DIM;
    int e  = sel[b * TOPK + kk];
    float w = wgt[b * TOPK + kk];
    BwT[id] = f2bf(w * Bl[((size_t)e * R_DIM + r) * H_DIM + n]);
  }
}

// ---------------- h_all[t][b][r'] = x[t,b,:] @ A'_b  (WMMA, N=16, double-buffered) ----------------
__global__ void __launch_bounds__(256)
lora_h_kernel(const float* __restrict__ x,
              const unsigned short* __restrict__ AbfT,
              float* __restrict__ h_all) {
  __shared__ __align__(16) unsigned short xs[2][128 * XS_STRIDE];
  __shared__ __align__(16) unsigned short at[2][16 * XS_STRIDE];
  int t0 = blockIdx.x * 128;
  int b  = blockIdx.y;
  int tid  = threadIdx.x;
  int lane = tid & 31, wave = tid >> 5;
  int ln16 = lane & 15;
  int koff_a = (lane < 16) ? 0 : 8;
  int koff_b = (lane < 16) ? 0 : 16;
  v8f_t acc = {};
  float4 xr[4];
  uint4  ar;

  auto load_regs = [&](int k0) {
#pragma unroll
    for (int it = 0; it < 4; ++it) {
      int i = tid + it * 256, m = i >> 3, q = i & 7;
      xr[it] = *(const float4*)&x[((size_t)(t0 + m) * B_DIM + b) * H_DIM + k0 + q * 4];
    }
    if (tid < 64) {
      int r = tid >> 2, q = tid & 3;
      ar = *(const uint4*)&AbfT[((size_t)b * KR + r) * H_DIM + k0 + q * 8];
    }
  };
  auto store_lds = [&](int buf) {
#pragma unroll
    for (int it = 0; it < 4; ++it) {
      int i = tid + it * 256, m = i >> 3, q = i & 7;
      *(uint2*)&xs[buf][m * XS_STRIDE + q * 4] =
          make_uint2(pk_bf16(xr[it].x, xr[it].y), pk_bf16(xr[it].z, xr[it].w));
    }
    if (tid < 64) {
      int r = tid >> 2, q = tid & 3;
      *(uint4*)&at[buf][r * XS_STRIDE + q * 8] = ar;
    }
  };
  auto compute = [&](int buf) {
    V16BF av, bv;
    int m = wave * 16 + ln16;
    av.u[0] = *(const uint4*)&xs[buf][m * XS_STRIDE + koff_a];
    av.u[1] = *(const uint4*)&xs[buf][m * XS_STRIDE + koff_a + 16];
    bv.u[0] = *(const uint4*)&at[buf][ln16 * XS_STRIDE + koff_a];
    bv.u[1] = *(const uint4*)&at[buf][ln16 * XS_STRIDE + koff_a + 16];
    acc = __builtin_amdgcn_wmma_f32_16x16x32_bf16(false, av.v, false, bv.v,
                                                  (short)0, acc, false, false);
  };

  load_regs(0);
  store_lds(0);
  __syncthreads();
  for (int kk = 0; kk < KSTEPS; kk += 2) {
    if (kk + 1 < KSTEPS) load_regs((kk + 1) * 32);
    compute(0);
    if (kk + 1 < KSTEPS) store_lds(1);
    __syncthreads();
    if (kk + 2 < KSTEPS) load_regs((kk + 2) * 32);
    if (kk + 1 < KSTEPS) compute(1);
    if (kk + 2 < KSTEPS) store_lds(0);
    __syncthreads();
  }
  int n = ln16;
  int mbase = wave * 16 + ((lane >= 16) ? 8 : 0);
#pragma unroll
  for (int i = 0; i < 8; ++i) {
    int m = mbase + i;
    h_all[((size_t)(t0 + m) * B_DIM + b) * KR + n] = acc[i];
  }
}

// ---------------- main: out = x@W + bias + h@B'w  (K-augmented, double-buffered WMMA) ----------------
__global__ void __launch_bounds__(256)
moe_main_kernel(const float* __restrict__ x,
                const unsigned short* __restrict__ WbfT,
                const float* __restrict__ bias,
                const unsigned short* __restrict__ BwT,
                const float* __restrict__ h_all,
                float* __restrict__ out) {
  __shared__ __align__(16) unsigned short xs[2][128 * XS_STRIDE];
  __shared__ __align__(16) unsigned short wt[2][128 * XS_STRIDE];
  int t0 = blockIdx.x * 128;
  int n0 = blockIdx.y * 128;
  int b  = blockIdx.z;
  int tid  = threadIdx.x;
  int lane = tid & 31, wave = tid >> 5;
  int ln16 = lane & 15;
  int wm = wave & 3;                    // M quadrant (32 rows)
  int wn = wave >> 2;                   // N half (64 cols)
  int koff_a = (lane < 16) ? 0 : 8;
  int koff_b = (lane < 16) ? 0 : 16;
  v8f_t acc[2][4] = {};
  float4 xr[4];
#ifndef HAVE_ASYNC_LDS
  uint4  wr[2];
#endif

  auto load_x_regs = [&](int k0) {
#pragma unroll
    for (int it = 0; it < 4; ++it) {
      int i = tid + it * 256, m = i >> 3, q = i & 7;
      xr[it] = *(const float4*)&x[((size_t)(t0 + m) * B_DIM + b) * H_DIM + k0 + q * 4];
    }
  };
  auto store_x_lds = [&](int buf) {
#pragma unroll
    for (int it = 0; it < 4; ++it) {
      int i = tid + it * 256, m = i >> 3, q = i & 7;
      *(uint2*)&xs[buf][m * XS_STRIDE + q * 4] =
          make_uint2(pk_bf16(xr[it].x, xr[it].y), pk_bf16(xr[it].z, xr[it].w));
    }
  };
#ifdef HAVE_ASYNC_LDS
  // CDNA5 async DMA: bf16 W tile goes global -> LDS without touching VGPRs.
  auto w_prefetch = [&](int k0, int buf) {
#pragma unroll
    for (int it = 0; it < 2; ++it) {
      int i = tid + it * 256, n = i >> 2, q = i & 3;
      __builtin_amdgcn_global_load_async_to_lds_b128(
          (v4i_t*)&WbfT[(size_t)(n0 + n) * H_DIM + k0 + q * 8],
          (v4i_t*)&wt[buf][n * XS_STRIDE + q * 8],
          0, 0);
    }
  };
#define W_COMMIT(buf)
#define ASYNC_FENCE() __builtin_amdgcn_s_wait_asynccnt(0)
#else
  auto w_prefetch = [&](int k0, int buf) {
    (void)buf;
#pragma unroll
    for (int it = 0; it < 2; ++it) {
      int i = tid + it * 256, n = i >> 2, q = i & 3;
      wr[it] = *(const uint4*)&WbfT[(size_t)(n0 + n) * H_DIM + k0 + q * 8];
    }
  };
#define W_COMMIT(buf)                                                        \
  {                                                                          \
    _Pragma("unroll")                                                        \
    for (int it = 0; it < 2; ++it) {                                         \
      int i = tid + it * 256, n = i >> 2, q = i & 3;                         \
      *(uint4*)&wt[buf][n * XS_STRIDE + q * 8] = wr[it];                     \
    }                                                                        \
  }
#define ASYNC_FENCE()
#endif
  auto compute = [&](int buf) {
    V16BF av[2], bv[4];
#pragma unroll
    for (int sm = 0; sm < 2; ++sm) {
      int m = wm * 32 + sm * 16 + ln16;
      av[sm].u[0] = *(const uint4*)&xs[buf][m * XS_STRIDE + koff_a];
      av[sm].u[1] = *(const uint4*)&xs[buf][m * XS_STRIDE + koff_a + 16];
    }
#pragma unroll
    for (int sn = 0; sn < 4; ++sn) {
      int n = wn * 64 + sn * 16 + ln16;
      bv[sn].u[0] = *(const uint4*)&wt[buf][n * XS_STRIDE + koff_a];
      bv[sn].u[1] = *(const uint4*)&wt[buf][n * XS_STRIDE + koff_a + 16];
    }
#pragma unroll
    for (int sm = 0; sm < 2; ++sm)
#pragma unroll
      for (int sn = 0; sn < 4; ++sn)
        acc[sm][sn] = __builtin_amdgcn_wmma_f32_16x16x32_bf16(
            false, av[sm].v, false, bv[sn].v, (short)0, acc[sm][sn], false, false);
  };

  // ---- pipelined main K loop ----
  load_x_regs(0);
  w_prefetch(0, 0);
  store_x_lds(0);
  W_COMMIT(0);
  ASYNC_FENCE();
  __syncthreads();
  for (int kk = 0; kk < KSTEPS; kk += 2) {
    if (kk + 1 < KSTEPS) { load_x_regs((kk + 1) * 32); w_prefetch((kk + 1) * 32, 1); }
    compute(0);
    if (kk + 1 < KSTEPS) { store_x_lds(1); W_COMMIT(1); }
    ASYNC_FENCE();
    __syncthreads();
    if (kk + 2 < KSTEPS) { load_x_regs((kk + 2) * 32); w_prefetch((kk + 2) * 32, 0); }
    if (kk + 1 < KSTEPS) compute(1);
    if (kk + 2 < KSTEPS) { store_x_lds(0); W_COMMIT(0); }
    ASYNC_FENCE();
    __syncthreads();
  }

  // ---- LoRA augmentation slice (K-pad 16 -> 32) into buffer 0 ----
#pragma unroll
  for (int it = 0; it < 8; ++it) {            // h values: 128 rows x 16 (coalesced)
    int i = tid + it * 256, m = i >> 4, k = i & 15;
    xs[0][m * XS_STRIDE + k] = f2bf(h_all[((size_t)(t0 + m) * B_DIM + b) * KR + k]);
  }
#pragma unroll
  for (int it = 0; it < 8; ++it) {            // zero pad K=16..31
    int i = tid + it * 256, m = i >> 4, k = 16 + (i & 15);
    xs[0][m * XS_STRIDE + k] = 0;
  }
#pragma unroll
  for (int it = 0; it < 8; ++it) {            // weighted B' rows (coalesced)
    int i = tid + it * 256, n = i >> 4, k = i & 15;
    wt[0][n * XS_STRIDE + k] = BwT[((size_t)b * H_DIM + n0 + n) * KR + k];
  }
#pragma unroll
  for (int it = 0; it < 8; ++it) {
    int i = tid + it * 256, n = i >> 4, k = 16 + (i & 15);
    wt[0][n * XS_STRIDE + k] = 0;
  }
  __syncthreads();
  compute(0);

  // ---- epilogue: + bias, store f32 ----
  int mb = (lane >= 16) ? 8 : 0;
#pragma unroll
  for (int sm = 0; sm < 2; ++sm) {
#pragma unroll
    for (int sn = 0; sn < 4; ++sn) {
      int n = n0 + wn * 64 + sn * 16 + ln16;
      float bn = bias[n];
#pragma unroll
      for (int i = 0; i < 8; ++i) {
        int m = t0 + wm * 32 + sm * 16 + mb + i;
        out[((size_t)m * B_DIM + b) * H_DIM + n] = acc[sm][sn][i] + bn;
      }
    }
  }
}

extern "C" void kernel_launch(void* const* d_in, const int* in_sizes, int n_in,
                              void* d_out, int out_size, void* d_ws, size_t ws_size,
                              hipStream_t stream) {
  (void)in_sizes; (void)n_in; (void)out_size; (void)ws_size;
  const float* x    = (const float*)d_in[0];
  const float* W    = (const float*)d_in[1];
  const float* bias = (const float*)d_in[2];
  const float* Wg   = (const float*)d_in[3];
  const float* A    = (const float*)d_in[4];
  const float* Bl   = (const float*)d_in[5];
  float* out = (float*)d_out;

  char* ws = (char*)d_ws;
  float*          pooled  = (float*)(ws + 0);                     // 128 KB
  int*            sel     = (int*)  (ws + 131072);                // 128 B
  float*          wgt     = (float*)(ws + 131072 + 256);          // 128 B
  unsigned short* WbfT    = (unsigned short*)(ws + (1u  << 20));  // 8 MB  [n][k] bf16
  unsigned short* AbfT    = (unsigned short*)(ws + (9u  << 20));  // 1 MB  [b][r'][h] bf16
  unsigned short* BwT     = (unsigned short*)(ws + (10u << 20));  // 1 MB  [b][n][r'] bf16
  float*          h_all   = (float*)(ws + (11u << 20));           // 2 MB  [t][b][r'] f32
  float*          partial = (float*)(ws + (13u << 20));           // 2 MB  pooling partials

  pool_partial_kernel<<<dim3(B_DIM * H_DIM / 256, T_DIM / TCHUNK), 256, 0, stream>>>(x, partial);
  pool_reduce_kernel<<<B_DIM * H_DIM / 256, 256, 0, stream>>>(partial, pooled);
  gate_kernel<<<1, 256, 0, stream>>>(pooled, Wg, sel, wgt);
  transposeW_kernel<<<dim3(H_DIM / 32, H_DIM / 32), 256, 0, stream>>>(W, WbfT);
  factors_kernel<<<B_DIM * KR * H_DIM / 256, 256, 0, stream>>>(A, Bl, sel, wgt, AbfT, BwT);
  lora_h_kernel<<<dim3(T_DIM / 128, B_DIM), 256, 0, stream>>>(x, AbfT, h_all);
  moe_main_kernel<<<dim3(T_DIM / 128, H_DIM / 128, B_DIM), 256, 0, stream>>>(
      x, WbfT, bias, BwT, h_all, out);
}
